// GNNModel_35167192220460
// MI455X (gfx1250) — matrix operand order repaired and checked by reading, not compiled, not run on hardware
//
#include <hip/hip_runtime.h>
#include <hip/hip_bf16.h>

// ---------------- problem constants (from reference) ----------------
constexpr int NN   = 50000;           // nodes
constexpr int NE   = 800000;          // edges
constexpr int EH   = NE + NN;         // edges + self loops
constexpr int INC  = 128;             // IN_CH
constexpr int HID  = 32;              // HIDDEN
constexpr int HDS  = 8;               // HEADS
constexpr int HC   = HDS * HID;       // 256
constexpr int OUTC = 128;             // OUT_CH
constexpr int NG   = 64;              // graphs

// ---------------- WMMA types ----------------
typedef __attribute__((ext_vector_type(16))) __bf16        v16bf;
typedef __attribute__((ext_vector_type(8)))  float         v8f;
typedef __attribute__((ext_vector_type(4)))  unsigned int  v4u;

union Frag16 { v4u q[2]; unsigned int d[8]; v16bf v; };

// ---------------- helpers ----------------
__device__ inline unsigned short f2bf(float f) {
  unsigned int u = __float_as_uint(f);
  unsigned int r = u + 0x7FFFu + ((u >> 16) & 1u);   // round-to-nearest-even
  return (unsigned short)(r >> 16);
}

__device__ inline void atomicMaxF(float* addr, float v) {
  unsigned int* ua = (unsigned int*)addr;
  unsigned int old = *ua;
  while (true) {
    float f = __uint_as_float(old);
    if (f >= v) break;
    unsigned int assumed = old;
    old = atomicCAS(ua, assumed, __float_as_uint(v));
    if (old == assumed) break;
  }
}

// ---------------- generic bf16 WMMA GEMM ----------------
// C[M,N] = A[M,K](bf16) * B[K,N] with B pre-transposed: Bt[N,K] bf16.
// Whole Bt staged in LDS (N*K/2 dwords) via b128 copies. Compile-time N/K:
// fully unrolled K loop, immediate LDS offsets. Each wave owns CT adjacent
// 16x16 column tiles sharing one A fragment -> CT independent WMMA chains.
// Requires: K % 32 == 0, M % 16 == 0, N % (16*CT) == 0.
template <int N, int K, int CT>
__global__ void gemm_bf16_wmma(const unsigned short* __restrict__ A,
                               const unsigned short* __restrict__ Bt,
                               const float* __restrict__ bias,
                               float* __restrict__ C,
                               int M, int applyRelu) {
  __shared__ unsigned int ldsb[(N * K) / 2];
  const int tid = threadIdx.x;
  {
    constexpr int nvec = (N * K) / 8;                  // v4u count
    v4u* dst = (v4u*)ldsb;
    const v4u* src = (const v4u*)Bt;
    #pragma unroll 4
    for (int i = tid; i < nvec; i += 256) dst[i] = src[i];
  }
  __syncthreads();

  const int wave = tid >> 5;
  const int lane = tid & 31;
  constexpr int ncg = N / (16 * CT);                   // column groups per row
  const int tileId = blockIdx.x * 8 + wave;
  const int total  = (M >> 4) * ncg;
  if (tileId >= total) return;                         // wave-uniform guard

  const int rowTile = tileId / ncg;
  const int cg      = tileId % ncg;
  const int half = lane >> 4;                          // 0: lanes 0-15, 1: 16-31
  const int l15  = lane & 15;
  const int mrow = (rowTile << 4) + l15;
  const int col0 = cg * (16 * CT) + l15;

  v8f acc[CT];
  #pragma unroll
  for (int j = 0; j < CT; ++j) acc[j] = (v8f){};

  const v4u* arow = (const v4u*)(A + (size_t)mrow * K + (half << 3));
  const v4u* lbase = (const v4u*)ldsb;

  #pragma unroll
  for (int kk = 0; kk < K; kk += 32) {
    Frag16 a;
    // A: lane half selects K {0..7,16..23} vs {8..15,24..31}
    a.q[0] = arow[(kk >> 3) + 0];
    a.q[1] = arow[(kk >> 3) + 2];
    const int kb = kk + (half << 3);
    #pragma unroll
    for (int j = 0; j < CT; ++j) {
      Frag16 b;
      const v4u* bp = lbase + (((col0 + 16 * j) * K + kb) >> 3);
      b.q[0] = bp[0];
      b.q[1] = bp[2];
      acc[j] = __builtin_amdgcn_wmma_f32_16x16x32_bf16(
          false, a.v, false, b.v, (short)0, acc[j], false, false);
    }
  }

  #pragma unroll
  for (int j = 0; j < CT; ++j) {
    const int col = col0 + 16 * j;
    const float bv = bias ? bias[col] : 0.0f;
    #pragma unroll
    for (int r = 0; r < 8; ++r) {
      int row = (rowTile << 4) + (half << 3) + r;
      float v = acc[j][r] + bv;
      if (applyRelu) v = fmaxf(v, 0.0f);
      C[(size_t)row * N + col] = v;
    }
  }
}

// ---------------- element-wise / conversion kernels ----------------
__global__ void k_fill(float* __restrict__ p, float v, int n) {
  int i = blockIdx.x * blockDim.x + threadIdx.x;
  if (i < n) p[i] = v;
}

// packed: 2 floats -> 1 dword of bf16 per thread (n must be even)
__global__ void k_cvt_bf16x2(const float* __restrict__ s,
                             unsigned int* __restrict__ d, int n2) {
  int i = blockIdx.x * blockDim.x + threadIdx.x;
  if (i < n2) {
    unsigned int lo = f2bf(s[2 * i]);
    unsigned int hi = f2bf(s[2 * i + 1]);
    d[i] = lo | (hi << 16);
  }
}

// src is K x N (row-major); dst is N x K bf16 (i.e. B transposed)
__global__ void k_transpose_bf16(const float* __restrict__ s,
                                 unsigned short* __restrict__ d, int K, int N) {
  int i = blockIdx.x * blockDim.x + threadIdx.x;
  if (i < K * N) {
    int k = i / N, n = i % N;
    d[n * K + k] = f2bf(s[i]);
  }
}

// a_s / a_d: [N, HEADS] attention logits
__global__ void k_att(const float* __restrict__ h,
                      const float* __restrict__ att_src,
                      const float* __restrict__ att_dst,
                      float* __restrict__ a_s, float* __restrict__ a_d) {
  int i = blockIdx.x * blockDim.x + threadIdx.x;
  if (i >= NN * HDS) return;
  int n = i >> 3, hd = i & 7;
  const float* hp = h + (size_t)n * HC + hd * HID;
  const float* as = att_src + hd * HID;
  const float* ad = att_dst + hd * HID;
  float ss = 0.f, ds = 0.f;
  #pragma unroll
  for (int c = 0; c < HID; ++c) { float v = hp[c]; ss += v * as[c]; ds += v * ad[c]; }
  a_s[i] = ss; a_d[i] = ds;
}

__device__ inline void edge_sd(const long long* __restrict__ ei, int e,
                               long long& s, long long& d) {
  if (e < NE) { s = ei[e]; d = ei[NE + e]; }
  else        { s = d = (long long)(e - NE); }        // self loop
}

__device__ inline float lrelu02(float v) { return v > 0.f ? v : 0.2f * v; }

__global__ void k_edge_max(const long long* __restrict__ ei,
                           const float* __restrict__ a_s,
                           const float* __restrict__ a_d,
                           float* __restrict__ mmax) {
  int i = blockIdx.x * blockDim.x + threadIdx.x;
  if (i >= EH * HDS) return;
  int e = i >> 3, hd = i & 7;
  long long s, d; edge_sd(ei, e, s, d);
  float v = lrelu02(a_s[s * HDS + hd] + a_d[d * HDS + hd]);
  atomicMaxF(&mmax[d * HDS + hd], v);
}

__global__ void k_edge_den(const long long* __restrict__ ei,
                           const float* __restrict__ a_s,
                           const float* __restrict__ a_d,
                           const float* __restrict__ mmax,
                           float* __restrict__ denom) {
  int i = blockIdx.x * blockDim.x + threadIdx.x;
  if (i >= EH * HDS) return;
  int e = i >> 3, hd = i & 7;
  long long s, d; edge_sd(ei, e, s, d);
  float v = lrelu02(a_s[s * HDS + hd] + a_d[d * HDS + hd]);
  atomicAdd(&denom[d * HDS + hd], __expf(v - mmax[d * HDS + hd]));
}

// one block per edge, one thread per channel; alpha computed once per head
__global__ void k_edge_msg(const long long* __restrict__ ei,
                           const float* __restrict__ a_s,
                           const float* __restrict__ a_d,
                           const float* __restrict__ mmax,
                           const float* __restrict__ denom,
                           const float* __restrict__ h,
                           float* __restrict__ gout) {
  __shared__ float salpha[HDS];
  const int e = blockIdx.x;
  const int c = threadIdx.x;
  long long s, d; edge_sd(ei, e, s, d);
  const float* hrow = h + (size_t)s * HC;
  __builtin_prefetch(hrow + c, 0, 0);                  // overlap h-row fetch
  if (c < HDS) {
    float v = lrelu02(a_s[s * HDS + c] + a_d[d * HDS + c]);
    salpha[c] = __expf(v - mmax[d * HDS + c]) / denom[d * HDS + c];
  }
  __syncthreads();
  atomicAdd(&gout[d * HC + c], salpha[c >> 5] * hrow[c]);
}

// gout = relu(gout + b_gat); z = gout  (init for GIN accumulation)
__global__ void k_bias_relu(float* __restrict__ gout,
                            const float* __restrict__ b_gat,
                            float* __restrict__ z) {
  int i = blockIdx.x * blockDim.x + threadIdx.x;
  if (i >= NN * HC) return;
  float v = fmaxf(gout[i] + b_gat[i & (HC - 1)], 0.0f);
  gout[i] = v; z[i] = v;
}

// GIN: z[dst] += gout[src] over original edges; one block per edge
__global__ void k_gin(const long long* __restrict__ ei,
                      const float* __restrict__ gout,
                      float* __restrict__ z) {
  int e = blockIdx.x;
  int c = threadIdx.x;
  long long s = ei[e], d = ei[NE + e];
  const float* grow = gout + (size_t)s * HC;
  __builtin_prefetch(grow + c, 0, 0);
  atomicAdd(&z[d * HC + c], grow[c]);
}

__global__ void k_pool_sum(const long long* __restrict__ batch,
                           const float* __restrict__ z2,
                           float* __restrict__ sums) {
  int i = blockIdx.x * blockDim.x + threadIdx.x;
  if (i >= NN * HID) return;
  int n = i >> 5, c = i & 31;
  atomicAdd(&sums[(int)batch[n] * HID + c], z2[i]);
}

__global__ void k_pool_cnt(const long long* __restrict__ batch,
                           float* __restrict__ counts) {
  int i = blockIdx.x * blockDim.x + threadIdx.x;
  if (i < NN) atomicAdd(&counts[(int)batch[i]], 1.0f);
}

__global__ void k_final(const float* __restrict__ sums,
                        const float* __restrict__ counts,
                        const float* __restrict__ Wf,
                        const float* __restrict__ bf_,
                        float* __restrict__ out) {
  int i = blockIdx.x * blockDim.x + threadIdx.x;
  if (i >= NG * OUTC) return;
  int g = i >> 7, o = i & 127;
  float inv = 1.0f / fmaxf(counts[g], 1.0f);
  float acc = bf_[o];
  #pragma unroll
  for (int c = 0; c < HID; ++c)
    acc += (sums[g * HID + c] * inv) * Wf[c * OUTC + o];
  out[i] = acc;
}

// ---------------- launcher ----------------
extern "C" void kernel_launch(void* const* d_in, const int* in_sizes, int n_in,
                              void* d_out, int out_size, void* d_ws, size_t ws_size,
                              hipStream_t stream) {
  (void)in_sizes; (void)n_in; (void)out_size; (void)ws_size;
  const float*     x       = (const float*)d_in[0];
  const long long* ei      = (const long long*)d_in[1];
  const long long* batch   = (const long long*)d_in[2];
  const float*     Wgat    = (const float*)d_in[3];
  const float*     att_src = (const float*)d_in[4];
  const float*     att_dst = (const float*)d_in[5];
  const float*     b_gat   = (const float*)d_in[6];
  const float*     W1      = (const float*)d_in[7];
  const float*     b1      = (const float*)d_in[8];
  const float*     W2      = (const float*)d_in[9];
  const float*     b2      = (const float*)d_in[10];
  const float*     Wf      = (const float*)d_in[11];
  const float*     bf_     = (const float*)d_in[12];
  float*           out     = (float*)d_out;

  // workspace carve-up (256B aligned)
  char* base = (char*)d_ws;
  size_t off = 0;
  auto alloc = [&](size_t bytes) -> char* {
    char* p = base + off;
    off = (off + bytes + 255) & ~(size_t)255;
    return p;
  };
  unsigned short* xb   = (unsigned short*)alloc((size_t)NN * INC * 2);
  unsigned short* Wgt  = (unsigned short*)alloc((size_t)HC * INC * 2);   // [256,128]
  float*          h    = (float*)alloc((size_t)NN * HC * 4);
  float*          a_s  = (float*)alloc((size_t)NN * HDS * 4);
  float*          a_d  = (float*)alloc((size_t)NN * HDS * 4);
  float*          mmax = (float*)alloc((size_t)NN * HDS * 4);
  float*          den  = (float*)alloc((size_t)NN * HDS * 4);
  float*          gout = (float*)alloc((size_t)NN * HC * 4);
  float*          z    = (float*)alloc((size_t)NN * HC * 4);
  unsigned short* W1t  = (unsigned short*)alloc((size_t)HID * HC * 2);   // [32,256]
  float*          z1   = (float*)alloc((size_t)NN * HID * 4);
  unsigned short* z1b  = (unsigned short*)alloc((size_t)NN * HID * 2);
  unsigned short* W2t  = (unsigned short*)alloc((size_t)HID * HID * 2);  // [32,32]
  float*          z2   = (float*)alloc((size_t)NN * HID * 4);
  float*          sums = (float*)alloc((size_t)NG * HID * 4);
  float*          cnts = (float*)alloc((size_t)NG * 4);
  unsigned short* zb   = (unsigned short*)h;   // alias: h is dead once z exists

  const int T = 256;
  auto blks = [](long long n, int t) { return (unsigned int)((n + t - 1) / t); };

  // 1. precision prep
  k_cvt_bf16x2<<<blks((long long)NN * INC / 2, T), T, 0, stream>>>(
      x, (unsigned int*)xb, NN * INC / 2);
  k_transpose_bf16<<<blks((long long)INC * HC, T), T, 0, stream>>>(Wgat, Wgt, INC, HC);
  k_transpose_bf16<<<blks((long long)HC * HID, T), T, 0, stream>>>(W1, W1t, HC, HID);
  k_transpose_bf16<<<blks((long long)HID * HID, T), T, 0, stream>>>(W2, W2t, HID, HID);

  // 2. init accumulators
  k_fill<<<blks((long long)NN * HDS, T), T, 0, stream>>>(mmax, -3.0e38f, NN * HDS);
  k_fill<<<blks((long long)NN * HDS, T), T, 0, stream>>>(den, 0.0f, NN * HDS);
  k_fill<<<blks((long long)NN * HC, T), T, 0, stream>>>(gout, 0.0f, NN * HC);
  k_fill<<<blks(NG * HID, T), T, 0, stream>>>(sums, 0.0f, NG * HID);
  k_fill<<<blks(NG, T), T, 0, stream>>>(cnts, 0.0f, NG);

  // 3. h = x @ W_gat  (WMMA 50000x256x128; 4 col tiles/wave -> 4 WMMA chains)
  gemm_bf16_wmma<HC, INC, 4>
      <<<blks((long long)(NN / 16) * (HC / 64), 8), T, 0, stream>>>(
          xb, Wgt, nullptr, h, NN, 0);

  // 4. attention logits
  k_att<<<blks((long long)NN * HDS, T), T, 0, stream>>>(h, att_src, att_dst, a_s, a_d);

  // 5. edge softmax (max / denom / message)
  k_edge_max<<<blks((long long)EH * HDS, T), T, 0, stream>>>(ei, a_s, a_d, mmax);
  k_edge_den<<<blks((long long)EH * HDS, T), T, 0, stream>>>(ei, a_s, a_d, mmax, den);
  k_edge_msg<<<EH, T, 0, stream>>>(ei, a_s, a_d, mmax, den, h, gout);

  // 6. bias + relu, init z
  k_bias_relu<<<blks((long long)NN * HC, T), T, 0, stream>>>(gout, b_gat, z);

  // 7. GIN aggregation
  k_gin<<<NE, T, 0, stream>>>(ei, gout, z);

  // 8. MLP: z1 = relu(z @ W1 + b1)   (WMMA, K=256, 2 col tiles/wave)
  k_cvt_bf16x2<<<blks((long long)NN * HC / 2, T), T, 0, stream>>>(
      z, (unsigned int*)zb, NN * HC / 2);
  gemm_bf16_wmma<HID, HC, 2>
      <<<blks((long long)(NN / 16) * (HID / 32), 8), T, 0, stream>>>(
          zb, W1t, b1, z1, NN, 1);

  // 9. z2 = relu(z1 @ W2 + b2)       (WMMA, K=32, 2 col tiles/wave)
  k_cvt_bf16x2<<<blks((long long)NN * HID / 2, T), T, 0, stream>>>(
      z1, (unsigned int*)z1b, NN * HID / 2);
  gemm_bf16_wmma<HID, HID, 2>
      <<<blks((long long)(NN / 16) * (HID / 32), 8), T, 0, stream>>>(
          z1b, W2t, b2, z2, NN, 1);

  // 10. mean pool + final linear
  k_pool_sum<<<blks((long long)NN * HID, T), T, 0, stream>>>(batch, z2, sums);
  k_pool_cnt<<<blks(NN, T), T, 0, stream>>>(batch, cnts);
  k_final<<<blks(NG * OUTC, T), T, 0, stream>>>(sums, cnts, Wf, bf_, out);
}
